// MultiHeadAttention_2345052144128
// MI455X (gfx1250) — matrix operand order repaired and checked by reading, not compiled, not run on hardware
//
#include <hip/hip_runtime.h>
#include <hip/hip_bf16.h>

// ---------------------------------------------------------------------------
// Types for CDNA5 WMMA (wave32): v_wmma_f32_16x16x32_bf16
// ---------------------------------------------------------------------------
typedef __attribute__((ext_vector_type(16))) __bf16 v16bf;
typedef __attribute__((ext_vector_type(2)))  __bf16 bf16x2;
typedef __attribute__((ext_vector_type(8)))  float  v8f;

__device__ __forceinline__ v8f wmma_bf16(v16bf a, v16bf b, v8f c) {
  // (neg_a, A, neg_b, B, c_mod, C, reuse_a, reuse_b)
  return __builtin_amdgcn_wmma_f32_16x16x32_bf16(false, a, false, b,
                                                 (short)0, c, false, false);
}

// ---------------------------------------------------------------------------
// CDNA5 async copy: global -> LDS, 16B per lane, tracked by ASYNCcnt.
// Generic pointers to __shared__ carry the LDS byte offset in addr[31:0]
// (flat-aperture rule), which is exactly what the VDST operand wants.
// ---------------------------------------------------------------------------
__device__ __forceinline__ void async_b128_to_lds(void* lds_ptr,
                                                  const void* gptr) {
  const unsigned lds_off = (unsigned)(uintptr_t)lds_ptr;
  asm volatile("global_load_async_to_lds_b128 %0, %1, off"
               :
               : "v"(lds_off), "v"(gptr)
               : "memory");
}
__device__ __forceinline__ void wait_async_le4() {
  asm volatile("s_wait_asynccnt 0x4" ::: "memory");
}
__device__ __forceinline__ void wait_async_0() {
  asm volatile("s_wait_asynccnt 0x0" ::: "memory");
}

// ---------------------------------------------------------------------------
// Fragment loaders from LDS, following ISA §7.12.2 VGPR layouts.
// A (16x32 bf16): lane r=lane&15 row M=r; half=lane>>4; VGPR v:
//                 K = (v/4)*16 + half*8 + (v%4)*2   (pairs contiguous in K)
// B (32x16 bf16): lane n=lane&15 col N=n; VGPR v: K = (v/4)*16 + half*8 + (v%4)*2
// C (16x16 f32):  N = lane&15 ; M = (lane>>4)*8 + v
// ---------------------------------------------------------------------------
__device__ __forceinline__ v16bf load_a_frag(const __bf16* __restrict__ tile16,
                                             int ld, int lane, int koff) {
  const int r = lane & 15, half = lane >> 4;
  const __bf16* rowp = tile16 + r * ld + koff;
  v16bf f;
#pragma unroll
  for (int v = 0; v < 8; ++v) {
    const int k = ((v >> 2) << 4) + (half << 3) + ((v & 3) << 1);
    bf16x2 p = *(const bf16x2*)(rowp + k);
    f[2 * v] = p[0];
    f[2 * v + 1] = p[1];
  }
  return f;
}

__device__ __forceinline__ v16bf load_b_frag(const __bf16* __restrict__ tile,
                                             int ld, int lane, int nbase,
                                             int koff) {
  const int n = lane & 15, half = lane >> 4;
  const __bf16* rowp = tile + (nbase + n) * ld + koff;
  v16bf f;
#pragma unroll
  for (int v = 0; v < 8; ++v) {
    bf16x2 p = *(const bf16x2*)(rowp + ((v >> 2) << 4) + (half << 3) + ((v & 3) << 1));
    f[2 * v] = p[0];
    f[2 * v + 1] = p[1];
  }
  return f;
}

// ---------------------------------------------------------------------------
// fp32 -> bf16 conversion (vectorized x4)
// ---------------------------------------------------------------------------
__global__ __launch_bounds__(256) void cvt_f32_bf16(
    const float* __restrict__ in, __bf16* __restrict__ out, int n) {
  int i = (blockIdx.x * 256 + threadIdx.x) * 4;
  if (i + 3 < n) {
    const float4 f = *(const float4*)(in + i);
    out[i + 0] = (__bf16)f.x;
    out[i + 1] = (__bf16)f.y;
    out[i + 2] = (__bf16)f.z;
    out[i + 3] = (__bf16)f.w;
  } else {
    for (; i < n; ++i) out[i] = (__bf16)in[i];
  }
}

// ---------------------------------------------------------------------------
// Tiled NT GEMM: C[M,N] = A[M,K] * B[N,K]^T + bias[N]
// Block 256 threads = 8 waves; block tile 128x128, K-step 32.
// Double-buffered LDS staging via GLOBAL_LOAD_ASYNC_TO_LDS_B128:
// stage i+1 is issued before waiting on stage i (ASYNCcnt in-order), so the
// copy of the next K-slice overlaps this slice's WMMA work.
// ---------------------------------------------------------------------------
constexpr int G_TM = 128, G_TN = 128, G_TK = 32;
constexpr int G_LDT = G_TK + 8; // 40 bf16 = 80 B rows: 16B-aligned, skewed

__device__ __forceinline__ void gemm_stage_async(
    const __bf16* __restrict__ A, const __bf16* __restrict__ Bm,
    __bf16* AsBuf, __bf16* BsBuf, int bm, int bn, int K, int k0, int tid) {
#pragma unroll
  for (int i = 0; i < 2; ++i) {
    const int q = tid + i * 256;
    const int row = q >> 2, c4 = q & 3;
    async_b128_to_lds(AsBuf + row * G_LDT + c4 * 8,
                      A + (size_t)(bm + row) * K + k0 + c4 * 8);
    async_b128_to_lds(BsBuf + row * G_LDT + c4 * 8,
                      Bm + (size_t)(bn + row) * K + k0 + c4 * 8);
  }
}

template <bool OUT_BF16>
__global__ __launch_bounds__(256) void gemm_nt_bf16(
    const __bf16* __restrict__ A,   // [M][K]
    const __bf16* __restrict__ Bm,  // [N][K]
    const float* __restrict__ bias, // [N]
    void* __restrict__ Cout, int M, int N, int K) {
  __shared__ __bf16 As[2][G_TM * G_LDT];
  __shared__ __bf16 Bs[2][G_TN * G_LDT];

  const int tid = threadIdx.x;
  const int lane = tid & 31, wave = tid >> 5;
  const int wm = wave >> 1, wn = wave & 1;
  const int bm = blockIdx.y * G_TM;
  const int bn = blockIdx.x * G_TN;

  v8f acc[2][4];
  const v8f vzero = {0.f, 0.f, 0.f, 0.f, 0.f, 0.f, 0.f, 0.f};
#pragma unroll
  for (int mi = 0; mi < 2; ++mi)
#pragma unroll
    for (int ni = 0; ni < 4; ++ni) acc[mi][ni] = vzero;

  const int nIter = K / G_TK;
  gemm_stage_async(A, Bm, As[0], Bs[0], bm, bn, K, 0, tid);

  for (int it = 0; it < nIter; ++it) {
    const int cur = it & 1;
    if (it + 1 < nIter) {
      // prefetch next stage into the other buffer (its last readers were
      // separated from us by the previous iteration's trailing barrier)
      gemm_stage_async(A, Bm, As[cur ^ 1], Bs[cur ^ 1], bm, bn, K,
                       (it + 1) * G_TK, tid);
      wait_async_le4(); // stage `it` resident; the 4 new ops stay in flight
    } else {
      wait_async_0();
    }
    __syncthreads();

    v16bf af[2], bf_[4];
#pragma unroll
    for (int mi = 0; mi < 2; ++mi)
      af[mi] =
          load_a_frag(As[cur] + (wm * 32 + mi * 16) * G_LDT, G_LDT, lane, 0);
#pragma unroll
    for (int ni = 0; ni < 4; ++ni)
      bf_[ni] = load_b_frag(Bs[cur], G_LDT, lane, wn * 64 + ni * 16, 0);
#pragma unroll
    for (int mi = 0; mi < 2; ++mi)
#pragma unroll
      for (int ni = 0; ni < 4; ++ni)
        acc[mi][ni] = wmma_bf16(af[mi], bf_[ni], acc[mi][ni]);
    __syncthreads();
  }

  // Epilogue: bias + store (C layout: n = lane&15, m = (lane>>4)*8 + v)
  const int nn = lane & 15, hg = lane >> 4;
#pragma unroll
  for (int mi = 0; mi < 2; ++mi)
#pragma unroll
    for (int ni = 0; ni < 4; ++ni) {
      const int col = bn + wn * 64 + ni * 16 + nn;
      const float bv = bias[col];
#pragma unroll
      for (int v = 0; v < 8; ++v) {
        const int row = bm + wm * 32 + mi * 16 + hg * 8 + v;
        const float val = acc[mi][ni][v] + bv;
        if constexpr (OUT_BF16)
          ((__bf16*)Cout)[(size_t)row * N + col] = (__bf16)val;
        else
          ((float*)Cout)[(size_t)row * N + col] = val;
      }
    }
}

// ---------------------------------------------------------------------------
// Flash attention (bf16 in/out, f32 accumulate + online softmax, exp2 domain).
// qkv layout: [B*S][3E], col = which*E + h*D + d  (matches reshape(B,S,3,H,D))
// out layout: [B*S][E],  col = h*D + d            (matches transpose->reshape)
// One block = 128 threads = 4 waves; block owns a 64-row q-tile of one (b,h);
// each wave owns 16 q-rows. Q fragments hoisted out of the k-loop.
// ---------------------------------------------------------------------------
__global__ __launch_bounds__(128) void flash_attn_bf16(
    const __bf16* __restrict__ qkv, __bf16* __restrict__ out) {
  constexpr int S = 2048, H = 16, D = 64, E = 1024, ROW = 3 * E;
  constexpr int QB = 64, KB = 64, NQB = S / QB;
  constexpr int LDT = D + 8; // 72 bf16 = 144 B rows (16B-aligned)
  // scores pre-multiplied by 1/sqrt(D) * log2(e); softmax runs in exp2 domain
  constexpr float SCL = 0.125f * 1.44269504088896340736f;

  __shared__ __bf16 Qs[QB * LDT];
  __shared__ __bf16 Ks[KB * LDT];
  __shared__ __bf16 Vts[D * LDT];    // transposed: [d][k]
  __shared__ __bf16 Ps[4][16 * LDT]; // per-wave P tile [16 q][64 k]

  const int tid = threadIdx.x, lane = tid & 31, wave = tid >> 5;
  const int qb = blockIdx.x % NQB;
  const int h = (blockIdx.x / NQB) % H;
  const int b = blockIdx.x / (NQB * H);
  const int nn = lane & 15, hg = lane >> 4;

  // Stage Q tile (64x64 bf16 = 512 x b128, 4 per thread).
  const __bf16* qbase = qkv + (size_t)(b * S + qb * QB) * ROW + h * D;
#pragma unroll
  for (int i = 0; i < 4; ++i) {
    const int q = tid + i * 128;
    const int row = q >> 3, c8 = q & 7;
    *(uint4*)(Qs + row * LDT + c8 * 8) =
        *(const uint4*)(qbase + (size_t)row * ROW + c8 * 8);
  }
  __syncthreads();
  // Q A-fragments are k-loop invariant: load once.
  const v16bf aq0 = load_a_frag(Qs + wave * 16 * LDT, LDT, lane, 0);
  const v16bf aq1 = load_a_frag(Qs + wave * 16 * LDT, LDT, lane, 32);

  float m_run[8], l_run[8];
  v8f acc_o[4];
  const v8f vzero = {0.f, 0.f, 0.f, 0.f, 0.f, 0.f, 0.f, 0.f};
#pragma unroll
  for (int v = 0; v < 8; ++v) { m_run[v] = -3.0e38f; l_run[v] = 0.f; }
#pragma unroll
  for (int di = 0; di < 4; ++di) acc_o[di] = vzero;

  for (int kb = 0; kb < S / KB; ++kb) {
    __syncthreads(); // previous compute done before restaging K/V
    const __bf16* kbase = qkv + (size_t)(b * S + kb * KB) * ROW + E + h * D;
    const __bf16* vbase =
        qkv + (size_t)(b * S + kb * KB) * ROW + 2 * E + h * D;
#pragma unroll
    for (int i = 0; i < 4; ++i) {
      const int q = tid + i * 128;
      const int row = q >> 3, c8 = q & 7;
      *(uint4*)(Ks + row * LDT + c8 * 8) =
          *(const uint4*)(kbase + (size_t)row * ROW + c8 * 8);
      // V staged transposed so PV B-fragment K-pairs are contiguous.
      uint4 raw = *(const uint4*)(vbase + (size_t)row * ROW + c8 * 8);
      __bf16 tmp[8];
      __builtin_memcpy(tmp, &raw, 16);
#pragma unroll
      for (int e = 0; e < 8; ++e) Vts[(c8 * 8 + e) * LDT + row] = tmp[e];
    }
    __syncthreads();

    // ---- S = Q K^T (16 q-rows x 64 k-cols per wave) ----
    v8f sacc[4];
#pragma unroll
    for (int ni = 0; ni < 4; ++ni) {
      sacc[ni] = vzero;
      v16bf bk0 = load_b_frag(Ks, LDT, lane, ni * 16, 0);
      sacc[ni] = wmma_bf16(aq0, bk0, sacc[ni]);
      v16bf bk1 = load_b_frag(Ks, LDT, lane, ni * 16, 32);
      sacc[ni] = wmma_bf16(aq1, bk1, sacc[ni]);
    }

    // ---- online softmax (row m = hg*8+v lives on one 16-lane half) ----
#pragma unroll
    for (int v = 0; v < 8; ++v) {
      float mx = m_run[v];
#pragma unroll
      for (int ni = 0; ni < 4; ++ni) mx = fmaxf(mx, sacc[ni][v] * SCL);
#pragma unroll
      for (int off = 1; off <= 8; off <<= 1)
        mx = fmaxf(mx, __shfl_xor(mx, off, 32));
      const float alpha = exp2f(m_run[v] - mx);
      float rs = 0.f;
#pragma unroll
      for (int ni = 0; ni < 4; ++ni) {
        const float p = exp2f(sacc[ni][v] * SCL - mx);
        sacc[ni][v] = p;
        rs += p;
      }
#pragma unroll
      for (int off = 1; off <= 8; off <<= 1) rs += __shfl_xor(rs, off, 32);
      m_run[v] = mx;
      l_run[v] = l_run[v] * alpha + rs;
#pragma unroll
      for (int di = 0; di < 4; ++di) acc_o[di][v] *= alpha;
    }

    // ---- stage P (bf16) into per-wave LDS to re-enter as A fragment ----
#pragma unroll
    for (int ni = 0; ni < 4; ++ni)
#pragma unroll
      for (int v = 0; v < 8; ++v)
        Ps[wave][(hg * 8 + v) * LDT + ni * 16 + nn] = (__bf16)sacc[ni][v];

    // ---- O += P V ----
    v16bf ap0 = load_a_frag(Ps[wave], LDT, lane, 0);
    v16bf ap1 = load_a_frag(Ps[wave], LDT, lane, 32);
#pragma unroll
    for (int di = 0; di < 4; ++di) {
      v16bf bv0 = load_b_frag(Vts, LDT, lane, di * 16, 0);
      acc_o[di] = wmma_bf16(ap0, bv0, acc_o[di]);
      v16bf bv1 = load_b_frag(Vts, LDT, lane, di * 16, 32);
      acc_o[di] = wmma_bf16(ap1, bv1, acc_o[di]);
    }
  }

  // ---- normalize + store to [B*S][E] (col = h*D + d) ----
  __bf16* obase = out + (size_t)(b * S + qb * QB + wave * 16) * E + h * D;
#pragma unroll
  for (int v = 0; v < 8; ++v) {
    const float inv = 1.0f / l_run[v];
    const int m = hg * 8 + v;
#pragma unroll
    for (int di = 0; di < 4; ++di)
      obase[(size_t)m * E + di * 16 + nn] = (__bf16)(acc_o[di][v] * inv);
  }
}

// ---------------------------------------------------------------------------
// Launcher
// ---------------------------------------------------------------------------
extern "C" void kernel_launch(void* const* d_in, const int* in_sizes, int n_in,
                              void* d_out, int out_size, void* d_ws,
                              size_t ws_size, hipStream_t stream) {
  (void)in_sizes; (void)n_in; (void)out_size; (void)ws_size;
  constexpr int B = 2, S = 2048, E = 1024;
  constexpr int M = B * S;   // 4096
  constexpr int N1 = 3 * E;  // 3072

  const float* x     = (const float*)d_in[0];
  const float* W_qkv = (const float*)d_in[1];
  const float* b_qkv = (const float*)d_in[2];
  const float* W_out = (const float*)d_in[3];
  const float* b_out = (const float*)d_in[4];
  float* out = (float*)d_out;

  char* ws = (char*)d_ws;
  __bf16* xb    = (__bf16*)ws; ws += (size_t)M * E * 2;
  __bf16* wqkvb = (__bf16*)ws; ws += (size_t)N1 * E * 2;
  __bf16* woutb = (__bf16*)ws; ws += (size_t)E * E * 2;
  __bf16* qkvb  = (__bf16*)ws; ws += (size_t)M * N1 * 2;
  __bf16* attnb = (__bf16*)ws;

  // 1) fp32 -> bf16 conversions
  {
    int n = M * E;
    cvt_f32_bf16<<<(n + 1023) / 1024, 256, 0, stream>>>(x, xb, n);
    n = N1 * E;
    cvt_f32_bf16<<<(n + 1023) / 1024, 256, 0, stream>>>(W_qkv, wqkvb, n);
    n = E * E;
    cvt_f32_bf16<<<(n + 1023) / 1024, 256, 0, stream>>>(W_out, woutb, n);
  }

  // 2) qkv = x @ W_qkv^T + b_qkv  -> bf16 [M][3E]
  gemm_nt_bf16<true><<<dim3(N1 / 128, M / 128), 256, 0, stream>>>(
      xb, wqkvb, b_qkv, qkvb, M, N1, E);

  // 3) flash attention -> bf16 [M][E]
  flash_attn_bf16<<<B * 16 * (S / 64), 128, 0, stream>>>(qkvb, attnb);

  // 4) out = attn @ W_out^T + b_out -> f32 [M][E]
  gemm_nt_bf16<false><<<dim3(E / 128, M / 128), 256, 0, stream>>>(
      attnb, woutb, b_out, out, M, E, E);
}
